// AttentionMIL_3023656976463
// MI455X (gfx1250) — matrix operand order repaired and checked by reading, not compiled, not run on hardware
//
#include <hip/hip_runtime.h>
#include <hip/hip_bf16.h>
#include <math.h>

typedef __bf16 bf16_t;
typedef __attribute__((ext_vector_type(16))) __bf16 v16bf;
typedef __attribute__((ext_vector_type(8)))  __bf16 v8bf;
typedef __attribute__((ext_vector_type(8)))  float  v8f;

#define NTOT 131072
#define DIN  1024
#define HIDN 256
#define ATTD 128
#define NBAG 64
#define SEGL (NTOT / NBAG)   // 2048, segment_ids = i/2048 (contiguous equal bags)
#define LNEPS 1e-5f

// ---------------------------------------------------------------------------
// helpers
// ---------------------------------------------------------------------------
__device__ __forceinline__ float gelu_exact(float x) {
  return 0.5f * x * (1.0f + erff(x * 0.70710678118654752f));
}

__device__ __forceinline__ v8f wmma_bf16(v16bf a, v16bf b, v8f c) {
  // D = A(16x32 bf16) * B(32x16 bf16) + C(16x16 f32)
  return __builtin_amdgcn_wmma_f32_16x16x32_bf16(
      /*neg_a=*/false, a, /*neg_b=*/false, b,
      /*c_mod=*/(short)0, c, /*reuse_a=*/false, /*reuse_b=*/false);
}

// A fragment from LDS, row-major [16][ld] bf16.
// lane<16 : row=l, K = kc*32 + {0..7, 16..23}
// lane>=16: row=l, K = kc*32 + {8..15, 24..31}
__device__ __forceinline__ v16bf load_a_frag(const bf16_t* src, int ld,
                                             int l, int half, int kc) {
  const bf16_t* p = src + l * ld + kc * 32 + half * 8;
  union { v16bf v; v8bf h[2]; } u;
  u.h[0] = *(const v8bf*)(p);
  u.h[1] = *(const v8bf*)(p + 16);
  return u.v;
}

// B fragment from pre-packed global weights: tile (nt,kc), 32 lanes x 16 halves
// contiguous -> one 32B load per lane.
__device__ __forceinline__ v16bf load_b_frag(const bf16_t* Wp, int KC,
                                             int nt, int kc, int lane) {
  return *(const v16bf*)(Wp + (((size_t)(nt * KC + kc) * 32 + lane) << 4));
}

// ---------------------------------------------------------------------------
// Weight packing: fp32 row-major [K][Ncols] -> bf16 WMMA-B fragment layout.
// B 32x16 layout: lane<16 holds col=lane, K = kc*32 + {0..7, 16..23};
//                 lane>=16 holds col=lane-16, K = kc*32 + {8..15, 24..31}.
// ---------------------------------------------------------------------------
__global__ __launch_bounds__(256) void mil_pack_w(const float* __restrict__ W,
                                                  bf16_t* __restrict__ out,
                                                  int K, int Ncols) {
  int gid   = blockIdx.x * 256 + threadIdx.x;
  int KC    = K >> 5;
  int tiles = (Ncols >> 4) * KC;
  int tile  = gid >> 5;
  if (tile >= tiles) return;
  int lane = gid & 31;
  int nt = tile / KC, kc = tile % KC;
  int l = lane & 15, half = lane >> 4;
  int col = nt * 16 + l;
  int kb  = kc * 32 + half * 8;
  bf16_t* o = out + ((size_t)gid << 4);
#pragma unroll
  for (int e = 0; e < 16; ++e)
    o[e] = (bf16_t)W[(size_t)(kb + (e & 7) + ((e >> 3) << 4)) * Ncols + col];
}

// ---------------------------------------------------------------------------
// LN + exact gelu over LDS rows; wave w handles rows 2w, 2w+1 (HIDN=256 cols,
// 8 elems/lane, wave32 shuffle reduction). Writes bf16 into A2 (next GEMM A)
// and optionally spills bf16 h2 to global.
// ---------------------------------------------------------------------------
__device__ __forceinline__ void ln_gelu_rows(const float* Hs, bf16_t* A2,
                                             const float* __restrict__ g,
                                             const float* __restrict__ be,
                                             int wave, int lane,
                                             bf16_t* __restrict__ Hgout,
                                             long row0) {
  for (int rr = 0; rr < 2; ++rr) {
    int m = wave * 2 + rr;
    float vals[8];
    float s = 0.f;
#pragma unroll
    for (int j = 0; j < 8; ++j) {
      vals[j] = Hs[m * HIDN + lane + j * 32];
      s += vals[j];
    }
#pragma unroll
    for (int d = 16; d >= 1; d >>= 1) s += __shfl_xor(s, d);
    float mu = s * (1.0f / HIDN);
    float v = 0.f;
#pragma unroll
    for (int j = 0; j < 8; ++j) { float dd = vals[j] - mu; v += dd * dd; }
#pragma unroll
    for (int d = 16; d >= 1; d >>= 1) v += __shfl_xor(v, d);
    float inv = rsqrtf(v * (1.0f / HIDN) + LNEPS);
#pragma unroll
    for (int j = 0; j < 8; ++j) {
      int n = lane + j * 32;
      float hn = (vals[j] - mu) * inv * g[n] + be[n];
      float ge = gelu_exact(hn);
      A2[m * HIDN + n] = (bf16_t)ge;
      if (Hgout) Hgout[(row0 + m) * (long)HIDN + n] = (bf16_t)ge;
    }
  }
}

// ---------------------------------------------------------------------------
// Fused per-instance pipeline: 16 rows per block, 8 waves.
//   GEMM1(1024->256) -> LN/gelu -> GEMM2(256->256) -> LN/gelu (spill h2)
//   -> GEMM3 (Wv & Wu, 256->128 each) -> gated score a[i]
// ---------------------------------------------------------------------------
__global__ __launch_bounds__(256) void mil_main(
    const float* __restrict__ x,
    const bf16_t* __restrict__ W1p, const float* __restrict__ b1,
    const float* __restrict__ g1,   const float* __restrict__ be1,
    const bf16_t* __restrict__ W2p, const float* __restrict__ b2,
    const float* __restrict__ g2,   const float* __restrict__ be2,
    const bf16_t* __restrict__ Wvp, const float* __restrict__ bv,
    const bf16_t* __restrict__ Wup, const float* __restrict__ bu,
    const float* __restrict__ Ww,   const float* __restrict__ bw,
    bf16_t* __restrict__ Hg,        float* __restrict__ aout) {
  __shared__ __align__(16) bf16_t As[16 * DIN];    // 32 KB : x tile bf16
  __shared__ __align__(16) float  Hs[16 * HIDN];   // 16 KB : fp32 GEMM out
  __shared__ __align__(16) bf16_t A2[16 * HIDN];   //  8 KB : activation bf16
  __shared__ float asum[16];

  const int tid  = threadIdx.x;
  const int wave = tid >> 5;
  const int lane = tid & 31;
  const int l    = lane & 15;
  const int half = lane >> 4;
  const long row0 = (long)blockIdx.x * 16;

  // ---- stage x tile (16 x 1024 fp32 -> bf16 LDS) -------------------------
  {
    const float4* xv = (const float4*)(x + row0 * DIN);
#pragma unroll 4
    for (int i = tid; i < 16 * DIN / 4; i += 256) {
      float4 v = xv[i];
      int o = i * 4;
      As[o + 0] = (bf16_t)v.x;
      As[o + 1] = (bf16_t)v.y;
      As[o + 2] = (bf16_t)v.z;
      As[o + 3] = (bf16_t)v.w;
    }
  }
  __syncthreads();

  // ---- GEMM1: K=1024 (32 chunks), wave handles col tiles 2w, 2w+1 --------
  v8f acc0 = {}, acc1 = {};
  for (int kc = 0; kc < DIN / 32; ++kc) {
    v16bf a  = load_a_frag(As, DIN, l, half, kc);
    v16bf bA = load_b_frag(W1p, DIN / 32, 2 * wave + 0, kc, lane);
    v16bf bB = load_b_frag(W1p, DIN / 32, 2 * wave + 1, kc, lane);
    acc0 = wmma_bf16(a, bA, acc0);
    acc1 = wmma_bf16(a, bB, acc1);
  }
  {
    int n0 = (2 * wave) * 16 + l, n1 = n0 + 16;
    float bb0 = b1[n0], bb1 = b1[n1];
#pragma unroll
    for (int r = 0; r < 8; ++r) {
      int m = r + half * 8;  // C/D layout: lanes 16-31 carry rows 8..15
      Hs[m * HIDN + n0] = acc0[r] + bb0;
      Hs[m * HIDN + n1] = acc1[r] + bb1;
    }
  }
  __syncthreads();

  ln_gelu_rows(Hs, A2, g1, be1, wave, lane, nullptr, row0);
  __syncthreads();

  // ---- GEMM2: K=256 (8 chunks) -------------------------------------------
  acc0 = (v8f){}; acc1 = (v8f){};
  for (int kc = 0; kc < HIDN / 32; ++kc) {
    v16bf a  = load_a_frag(A2, HIDN, l, half, kc);
    v16bf bA = load_b_frag(W2p, HIDN / 32, 2 * wave + 0, kc, lane);
    v16bf bB = load_b_frag(W2p, HIDN / 32, 2 * wave + 1, kc, lane);
    acc0 = wmma_bf16(a, bA, acc0);
    acc1 = wmma_bf16(a, bB, acc1);
  }
  {
    int n0 = (2 * wave) * 16 + l, n1 = n0 + 16;
    float bb0 = b2[n0], bb1 = b2[n1];
#pragma unroll
    for (int r = 0; r < 8; ++r) {
      int m = r + half * 8;
      Hs[m * HIDN + n0] = acc0[r] + bb0;
      Hs[m * HIDN + n1] = acc1[r] + bb1;
    }
  }
  if (tid < 16) asum[tid] = 0.f;
  __syncthreads();  // waves done reading A2 + Hs fully written

  ln_gelu_rows(Hs, A2, g2, be2, wave, lane, Hg, row0);  // spill h2 (bf16)
  __syncthreads();

  // ---- GEMM3: t = h2*Wv, u = h2*Wu ; wave handles ATT col tile = wave ----
  v8f tac = {}, uac = {};
  for (int kc = 0; kc < HIDN / 32; ++kc) {
    v16bf a  = load_a_frag(A2, HIDN, l, half, kc);
    v16bf bV = load_b_frag(Wvp, HIDN / 32, wave, kc, lane);
    v16bf bU = load_b_frag(Wup, HIDN / 32, wave, kc, lane);
    tac = wmma_bf16(a, bV, tac);
    uac = wmma_bf16(a, bU, uac);
  }
  {
    int col = wave * 16 + l;  // 0..127
    float bvv = bv[col], buu = bu[col], www = Ww[col];
#pragma unroll
    for (int r = 0; r < 8; ++r) {
      float t = tanhf(tac[r] + bvv);
      float u = 1.0f / (1.0f + expf(-(uac[r] + buu)));
      float gte = t * u * www;
      // reduce over the 16 cols of this tile (xor stays within half)
      gte += __shfl_xor(gte, 1);
      gte += __shfl_xor(gte, 2);
      gte += __shfl_xor(gte, 4);
      gte += __shfl_xor(gte, 8);
      if (l == 0) atomicAdd(&asum[r + half * 8], gte);  // ds_add_f32
    }
  }
  __syncthreads();
  if (tid < 16) aout[row0 + tid] = asum[tid] + bw[0];
}

// ---------------------------------------------------------------------------
// Per-bag softmax over 2048 contiguous scores + attention-weighted h2 sum.
// ---------------------------------------------------------------------------
__global__ __launch_bounds__(256) void mil_softmax_pool(
    const float* __restrict__ a, const bf16_t* __restrict__ Hg,
    float* __restrict__ zbuf) {
  __shared__ float ebuf[SEGL];  // 8 KB
  __shared__ float red[8];
  const int tid = threadIdx.x, lane = tid & 31, wave = tid >> 5;
  const long base = (long)blockIdx.x * SEGL;

  float mx = -3.4e38f;
  for (int i = tid; i < SEGL; i += 256) mx = fmaxf(mx, a[base + i]);
#pragma unroll
  for (int d = 16; d >= 1; d >>= 1) mx = fmaxf(mx, __shfl_xor(mx, d));
  if (lane == 0) red[wave] = mx;
  __syncthreads();
  mx = red[0];
#pragma unroll
  for (int w = 1; w < 8; ++w) mx = fmaxf(mx, red[w]);
  __syncthreads();

  float s = 0.f;
  for (int i = tid; i < SEGL; i += 256) {
    float e = expf(a[base + i] - mx);
    ebuf[i] = e;
    s += e;
  }
#pragma unroll
  for (int d = 16; d >= 1; d >>= 1) s += __shfl_xor(s, d);
  if (lane == 0) red[wave] = s;
  __syncthreads();
  s = 0.f;
#pragma unroll
  for (int w = 0; w < 8; ++w) s += red[w];
  const float inv = 1.0f / s;

  // thread owns column tid; coalesced bf16 loads, LDS broadcast of weights
  float z = 0.f;
  const bf16_t* hp = Hg + base * HIDN + tid;
  for (int i = 0; i < SEGL; ++i)
    z = fmaf(ebuf[i], (float)hp[(size_t)i * HIDN], z);
  zbuf[blockIdx.x * HIDN + tid] = z * inv;
}

// ---------------------------------------------------------------------------
// Head: logits = gelu(z*Wc1 + bc1) * Wc2 + bc2  (64x256 -> 64x128 -> 64x2)
// ---------------------------------------------------------------------------
__global__ __launch_bounds__(128) void mil_head(
    const float* __restrict__ zbuf, const float* __restrict__ Wc1,
    const float* __restrict__ bc1,  const float* __restrict__ Wc2,
    const float* __restrict__ bc2,  float* __restrict__ out) {
  __shared__ float c1[ATTD];  // HID/2 = 128
  const int row = blockIdx.x, j = threadIdx.x;
  float s = bc1[j];
  for (int k = 0; k < HIDN; ++k)
    s = fmaf(zbuf[row * HIDN + k], Wc1[k * (HIDN / 2) + j], s);
  c1[j] = gelu_exact(s);
  __syncthreads();
  if (j < 2) {
    float o = bc2[j];
    for (int k = 0; k < HIDN / 2; ++k) o = fmaf(c1[k], Wc2[k * 2 + j], o);
    out[row * 2 + j] = o;
  }
}

// ---------------------------------------------------------------------------
extern "C" void kernel_launch(void* const* d_in, const int* in_sizes, int n_in,
                              void* d_out, int out_size, void* d_ws,
                              size_t ws_size, hipStream_t stream) {
  const float* x   = (const float*)d_in[0];
  // d_in[1]: segment_ids — unused, bags are contiguous blocks of 2048
  const float* W1  = (const float*)d_in[2];
  const float* b1  = (const float*)d_in[3];
  const float* g1  = (const float*)d_in[4];
  const float* be1 = (const float*)d_in[5];
  const float* W2  = (const float*)d_in[6];
  const float* b2  = (const float*)d_in[7];
  const float* g2  = (const float*)d_in[8];
  const float* be2 = (const float*)d_in[9];
  const float* Wv  = (const float*)d_in[10];
  const float* bv  = (const float*)d_in[11];
  const float* Wu  = (const float*)d_in[12];
  const float* bu  = (const float*)d_in[13];
  const float* Ww  = (const float*)d_in[14];
  const float* bw  = (const float*)d_in[15];
  const float* Wc1 = (const float*)d_in[16];
  const float* bc1 = (const float*)d_in[17];
  const float* Wc2 = (const float*)d_in[18];
  const float* bc2 = (const float*)d_in[19];

  char* ws = (char*)d_ws;
  size_t o = 0;
  bf16_t* W1p = (bf16_t*)(ws + o); o += (size_t)DIN * HIDN * 2;   // 512 KB
  bf16_t* W2p = (bf16_t*)(ws + o); o += (size_t)HIDN * HIDN * 2;  // 128 KB
  bf16_t* Wvp = (bf16_t*)(ws + o); o += (size_t)HIDN * ATTD * 2;  //  64 KB
  bf16_t* Wup = (bf16_t*)(ws + o); o += (size_t)HIDN * ATTD * 2;  //  64 KB
  float*  asc = (float*)(ws + o);  o += (size_t)NTOT * 4;         // 512 KB
  float*  zbf = (float*)(ws + o);  o += (size_t)NBAG * HIDN * 4;  //  64 KB
  bf16_t* Hg  = (bf16_t*)(ws + o); o += (size_t)NTOT * HIDN * 2;  //  64 MB

  // 1) pack weights into WMMA B-fragment bf16 layout (tiny, L2-resident)
  mil_pack_w<<<64, 256, 0, stream>>>(W1, W1p, DIN, HIDN);
  mil_pack_w<<<16, 256, 0, stream>>>(W2, W2p, HIDN, HIDN);
  mil_pack_w<<<8, 256, 0, stream>>>(Wv, Wvp, HIDN, ATTD);
  mil_pack_w<<<8, 256, 0, stream>>>(Wu, Wup, HIDN, ATTD);

  // 2) fused backbone + attention scores (the heavy pass: streams x once)
  mil_main<<<NTOT / 16, 256, 0, stream>>>(x, W1p, b1, g1, be1, W2p, b2, g2,
                                          be2, Wvp, bv, Wup, bu, Ww, bw, Hg,
                                          asc);

  // 3) per-bag softmax + attention pooling
  mil_softmax_pool<<<NBAG, 256, 0, stream>>>(asc, Hg, zbf);

  // 4) classifier head
  mil_head<<<NBAG, 128, 0, stream>>>(zbf, Wc1, bc1, Wc2, bc2, (float*)d_out);
}
